// GraphConv_20401094656620
// MI455X (gfx1250) — matrix operand order repaired and checked by reading, not compiled, not run on hardware
//
#include <hip/hip_runtime.h>
#include <hip/hip_bf16.h>

typedef __bf16 bf16_t;
typedef __attribute__((ext_vector_type(16))) __bf16 v16bf;
typedef __attribute__((ext_vector_type(8)))  float  v8f;
typedef __attribute__((ext_vector_type(4)))  float  v4f;

#define NN   16384      // N nodes
#define DD   64         // feature dim (in == out)
#define RCH  64         // row chunks for deterministic column-sum
#define RPC  (NN / RCH) // rows per chunk = 256

// ---------------------------------------------------------------------------
// Pass 1a: deterministic partial column sums of A (streams A once, NT loads)
// grid (NN/256, RCH) x 256 threads; thread owns one column within one chunk.
// ---------------------------------------------------------------------------
__global__ __launch_bounds__(256) void colsum_partial_kernel(
    const float* __restrict__ A, float* __restrict__ part) {
  const int col = blockIdx.x * 256 + threadIdx.x;
  const float* p = A + (size_t)(blockIdx.y * RPC) * NN + col;
  float s0 = 0.f, s1 = 0.f, s2 = 0.f, s3 = 0.f;
  #pragma unroll 4
  for (int r = 0; r < RPC; r += 4) {
    s0 += __builtin_nontemporal_load(p + (size_t)(r + 0) * NN);
    s1 += __builtin_nontemporal_load(p + (size_t)(r + 1) * NN);
    s2 += __builtin_nontemporal_load(p + (size_t)(r + 2) * NN);
    s3 += __builtin_nontemporal_load(p + (size_t)(r + 3) * NN);
  }
  part[(size_t)blockIdx.y * NN + col] = (s0 + s1) + (s2 + s3);
}

// ---------------------------------------------------------------------------
// Pass 1b: fixed-order reduction of the 64 partials -> colsum[col]
// ---------------------------------------------------------------------------
__global__ __launch_bounds__(256) void colsum_reduce_kernel(
    const float* __restrict__ part, float* __restrict__ colsum) {
  const int col = blockIdx.x * 256 + threadIdx.x;
  float s = 0.f;
  #pragma unroll
  for (int p = 0; p < RCH; ++p) s += part[(size_t)p * NN + col];
  colsum[col] = s;
}

// ---------------------------------------------------------------------------
// Pass 2: Y = (diag(1/(colsum+1)) * X) @ W, stored hi/lo bf16 directly in the
// WMMA B-operand (32x16 bf16) register layout:
//   fragment f = (kblock*4 + ntile); element (lane, e) with
//   lane = n + 16*(kk>=16), e = kk&15   (kk = row index within 32-block)
// One block per graph row i; 64 threads = 64 output columns.
// ---------------------------------------------------------------------------
__global__ __launch_bounds__(64) void y_build_kernel(
    const float* __restrict__ feat, const float* __restrict__ W,
    const float* __restrict__ colsum,
    bf16_t* __restrict__ yhi, bf16_t* __restrict__ ylo) {
  const int i = blockIdx.x;       // row 0..NN-1
  const int j = threadIdx.x;      // col 0..63
  const float dinv = 1.0f / (colsum[i] + 1.0f);

  __shared__ float xs[DD];
  xs[j] = feat[(size_t)i * DD + j] * dinv;
  __syncthreads();

  float acc = 0.f;
  #pragma unroll
  for (int c = 0; c < DD; ++c) acc += xs[c] * W[c * DD + j];

  const int kb = i >> 5, kk = i & 31;
  const int t = j >> 4, n = j & 15;
  const int lane = n + (kk & 16);
  const int e = kk & 15;
  const size_t off = ((size_t)(kb * 4 + t) * 32 + lane) * 16 + e;

  const bf16_t h = (bf16_t)acc;
  yhi[off] = h;
  ylo[off] = (bf16_t)(acc - (float)h);
}

// ---------------------------------------------------------------------------
// Pass 3: Z = A @ Y with bf16 hi/lo split-precision WMMA; fused epilogue
//   out[i][j] = relu( dinv[i] * (Z[i][j] + Y[i][j]) + bias[j] )
// 1 wave -> 16 rows x 64 cols. grid = NN/64 blocks x 128 threads (4 waves).
// A streamed nontemporal (1 GiB, can't fit L2); Y fragments are L2-resident.
// ---------------------------------------------------------------------------
__global__ __launch_bounds__(128) void gcn_main_kernel(
    const float* __restrict__ A,
    const bf16_t* __restrict__ yhi, const bf16_t* __restrict__ ylo,
    const float* __restrict__ colsum, const float* __restrict__ bias,
    float* __restrict__ out) {
  const int lane = threadIdx.x & 31;
  const int wave = threadIdx.x >> 5;
  const int rowBase = blockIdx.x * 64 + wave * 16;

  v8f acc[4] = {};  // 4 N-tiles of 16x16 f32

  // A-operand addressing (16x32 A, contiguous k order: K = khalf + e, same as B)
  const int m = lane & 15;
  const int khalf = (lane < 16) ? 0 : 16;
  const float* arow = A + (size_t)(rowBase + m) * NN;

  for (int kb = 0; kb < NN / 32; ++kb) {
    const int k0 = kb * 32 + khalf;
    // 16 A elements per lane: K = khalf + {0..15}
    v4f a0 = __builtin_nontemporal_load((const v4f*)(arow + k0));
    v4f a1 = __builtin_nontemporal_load((const v4f*)(arow + k0 + 4));
    v4f a2 = __builtin_nontemporal_load((const v4f*)(arow + k0 + 8));
    v4f a3 = __builtin_nontemporal_load((const v4f*)(arow + k0 + 12));

    float av[16];
    #pragma unroll
    for (int q = 0; q < 4; ++q) {
      av[q] = a0[q]; av[4 + q] = a1[q]; av[8 + q] = a2[q]; av[12 + q] = a3[q];
    }
    v16bf ahi, alo;
    #pragma unroll
    for (int e = 0; e < 16; ++e) {
      const float x = av[e];
      const bf16_t h = (bf16_t)x;
      ahi[e] = h;
      alo[e] = (bf16_t)(x - (float)h);
    }

    // B fragments (prebuilt layout): contiguous 32B per lane, L2-hot
    const size_t fb = (size_t)kb * 2048 + (size_t)lane * 16;
    v16bf bh[4], bl[4];
    #pragma unroll
    for (int t = 0; t < 4; ++t) {
      bh[t] = *(const v16bf*)(yhi + fb + t * 512);
      bl[t] = *(const v16bf*)(ylo + fb + t * 512);
    }

    // 3-product split accumulation; rotate accumulators to avoid RAW hazards
    #pragma unroll
    for (int t = 0; t < 4; ++t)
      acc[t] = __builtin_amdgcn_wmma_f32_16x16x32_bf16(
          false, ahi, false, bh[t], (short)0, acc[t], false, false);
    #pragma unroll
    for (int t = 0; t < 4; ++t)
      acc[t] = __builtin_amdgcn_wmma_f32_16x16x32_bf16(
          false, alo, false, bh[t], (short)0, acc[t], false, false);
    #pragma unroll
    for (int t = 0; t < 4; ++t)
      acc[t] = __builtin_amdgcn_wmma_f32_16x16x32_bf16(
          false, ahi, false, bl[t], (short)0, acc[t], false, false);
  }

  // Epilogue: C/D layout -> VGPR r: M = r + 8*(lane>=16); N = (lane&15)+16t
  const int nloc = lane & 15;
  #pragma unroll
  for (int r = 0; r < 8; ++r) {
    const int row = rowBase + r + ((lane >> 4) << 3);
    const float dinv = 1.0f / (colsum[row] + 1.0f);
    const int kb = row >> 5, kk = row & 31;
    const int flane = nloc + (kk & 16);
    const int e = kk & 15;
    #pragma unroll
    for (int t = 0; t < 4; ++t) {
      const int col = t * 16 + nloc;
      const size_t off = ((size_t)(kb * 4 + t) * 32 + flane) * 16 + e;
      const float y = (float)yhi[off] + (float)ylo[off];   // Y[row][col]
      const float v = dinv * (acc[t][r] + y) + bias[col];
      out[(size_t)row * DD + col] = v > 0.f ? v : 0.f;
    }
  }
}

// ---------------------------------------------------------------------------
// Workspace layout (8.07 MB):
//   [0, 4MB)            float part[64][16384]
//   [4MB, 4MB+64KB)     float colsum[16384]
//   [+0, +2MB)          bf16 Y_hi fragments
//   [+2MB, +4MB)        bf16 Y_lo fragments
// ---------------------------------------------------------------------------
extern "C" void kernel_launch(void* const* d_in, const int* in_sizes, int n_in,
                              void* d_out, int out_size, void* d_ws, size_t ws_size,
                              hipStream_t stream) {
  const float* A    = (const float*)d_in[0];
  const float* feat = (const float*)d_in[1];
  const float* W    = (const float*)d_in[2];
  const float* bias = (const float*)d_in[3];
  float* out = (float*)d_out;

  char* ws = (char*)d_ws;
  float*  part   = (float*)(ws);
  float*  colsum = (float*)(ws + (size_t)4 * 1024 * 1024);
  bf16_t* yhi    = (bf16_t*)(ws + (size_t)4 * 1024 * 1024 + 64 * 1024);
  bf16_t* ylo    = (bf16_t*)(ws + (size_t)4 * 1024 * 1024 + 64 * 1024 +
                             (size_t)2 * 1024 * 1024);

  colsum_partial_kernel<<<dim3(NN / 256, RCH), 256, 0, stream>>>(A, part);
  colsum_reduce_kernel<<<NN / 256, 256, 0, stream>>>(part, colsum);
  y_build_kernel<<<NN, DD, 0, stream>>>(feat, W, colsum, yhi, ylo);
  gcn_main_kernel<<<NN / 64, 128, 0, stream>>>(A, yhi, ylo, colsum, bias, out);
}
